// DynamicPointingDecoder_42004780155379
// MI455X (gfx1250) — compile-verified, matrix operand order checked by reading, not run on hardware
//
#include <hip/hip_runtime.h>
#include <hip/hip_bf16.h>
#include <math.h>

// Problem constants (from reference)
constexpr int NB    = 32;            // batch
constexpr int NT    = 1024;          // sequence length
constexpr int NH    = 128;           // H
constexpr int ND    = 256;           // D = 2H
constexpr int NROWS = NB * NT;       // 32768 flattened rows
constexpr int K1    = 384;           // 3H : concat(U, r)
constexpr int NPRE  = 2048;          // H * POOL pre-pool columns
constexpr int NITER = 4;

typedef __bf16 bf16;
typedef __attribute__((ext_vector_type(16))) __bf16 v16bf;
typedef __attribute__((ext_vector_type(8)))  float  v8f;

__device__ __forceinline__ bf16 f2bf(float f) {
  union { float f; unsigned u; } in; in.f = f;
  unsigned r = in.u + 0x7FFFu + ((in.u >> 16) & 1u);   // round-to-nearest-even
  union { unsigned short s; bf16 b; } out; out.s = (unsigned short)(r >> 16);
  return out.b;
}

// Load a 16-element bf16 WMMA fragment: two contiguous 16B chunks at +0 and +16
// elements (matches 16-bit A/B VGPR layout: elems 0..7 -> K=kb+khalf..+7,
// elems 8..15 -> K=kb+khalf+16..+23).
__device__ __forceinline__ v16bf load16(const bf16* p) {
  union { uint4 u[2]; v16bf v; } c;
  c.u[0] = *reinterpret_cast<const uint4*>(p);
  c.u[1] = *reinterpret_cast<const uint4*>(p + 16);
  return c.v;
}

// ---------------------------------------------------------------------------
// Fused GEMM + bias + maxout(16): out[M, NPRE/16] = maxpool16(A @ Bt^T + bias)
// A: [M, K] bf16 row-major (lda), Bt: [NPRE, K] bf16 row-major (N-major!).
// Grid: (M/128, NPRE/128), block 256 (8 wave32s). Wave w -> rows [w*16, +16).
// Each wave: 8 accumulators = 16 rows x 128 pre-pool columns; each 16x16 WMMA
// tile is exactly one maxout pool group -> cross-lane max in epilogue.
// LDS B-tile is double-buffered. sched_barrier fences force: [all fragment
// loads] -> single wait -> [8 back-to-back v_wmma], keeping the XDL pipe full
// instead of the ds_load/wait/wmma ping-pong the scheduler otherwise emits.
// ---------------------------------------------------------------------------
__global__ __launch_bounds__(256) void k_gemm_maxout(
    const bf16* __restrict__ A, int lda,
    const bf16* __restrict__ Bt, int ldb, int K,
    const float* __restrict__ bias,
    float* __restrict__ outF, bf16* __restrict__ outB, int ldo)
{
  __shared__ bf16 Bs[2][128 * 32];              // 2 x 8 KB B tiles
  const int lane  = threadIdx.x & 31;
  const int wave  = threadIdx.x >> 5;
  const int ncol  = lane & 15;                  // N index (B/C) and M index (A)
  const int khalf = (lane >> 4) << 3;           // 0 or 8: K sub-offset per half
  const int mbase = blockIdx.x * 128 + wave * 16;
  const int nbase = blockIdx.y * 128;

  // staging geometry: 128 rows x 64B per tile, 256 threads x 32B
  const int rr = threadIdx.x >> 1;
  const int hh = (threadIdx.x & 1) << 4;
  const bf16* stage_src = Bt + (size_t)(nbase + rr) * ldb + hh;

  v8f acc[8] = {};
  const bf16* Arow = A + (size_t)(mbase + ncol) * lda + khalf;

  // prologue: stage tile 0 into buffer 0
  {
    const uint4* s = reinterpret_cast<const uint4*>(stage_src);
    uint4* d = reinterpret_cast<uint4*>(&Bs[0][rr * 32 + hh]);
    d[0] = s[0]; d[1] = s[1];
  }

  int ib = 0;
  for (int kb = 0; kb < K; kb += 32, ib ^= 1) {
    __syncthreads();                            // tile kb visible in Bs[ib]

    // 1) issue global prefetch of tile kb+32 (no wait yet)
    uint4 pre0, pre1;
    const bool have_next = (kb + 32) < K;
    if (have_next) {
      const uint4* s = reinterpret_cast<const uint4*>(stage_src + kb + 32);
      pre0 = s[0]; pre1 = s[1];
    }

    // 2) load ALL fragments of the current tile first...
    const bf16* buf = &Bs[ib][0];
    v16bf bfrag[8];
#pragma unroll
    for (int t = 0; t < 8; ++t)
      bfrag[t] = load16(&buf[(t * 16 + ncol) * 32 + khalf]);
    v16bf afrag = load16(Arow + kb);

    // ...then fence the scheduler so no load is sunk between the WMMAs:
    __builtin_amdgcn_sched_barrier(0);
#pragma unroll
    for (int t = 0; t < 8; ++t)
      acc[t] = __builtin_amdgcn_wmma_f32_16x16x32_bf16(
          false, afrag, false, bfrag[t], (short)0, acc[t], false, false);
    __builtin_amdgcn_sched_barrier(0);

    // 3) commit the prefetched tile into the other LDS buffer
    if (have_next) {
      uint4* d = reinterpret_cast<uint4*>(&Bs[ib ^ 1][rr * 32 + hh]);
      d[0] = pre0; d[1] = pre1;
    }
  }

  // Epilogue: bias + maxpool over the 16 columns of each tile.
  // C layout: row m = r + 8*(lane>>4) in VGPR r; cols spread over lane&15.
#pragma unroll
  for (int t = 0; t < 8; ++t) {
    const float cb    = bias[nbase + t * 16 + ncol];
    const int outcol  = (nbase >> 4) + t;
#pragma unroll
    for (int r = 0; r < 8; ++r) {
      float v = acc[t][r] + cb;
      v = fmaxf(v, __shfl_xor(v, 1));
      v = fmaxf(v, __shfl_xor(v, 2));
      v = fmaxf(v, __shfl_xor(v, 4));
      v = fmaxf(v, __shfl_xor(v, 8));          // max over 16-lane half
      if (ncol == 0) {
        const int row = mbase + khalf + r;     // khalf == 8*(lane>>4)
        outF[(size_t)row * ldo + outcol] = v;
        if (outB) outB[(size_t)row * ldo + outcol] = f2bf(v);
      }
    }
  }
}

// ---------------------------------------------------------------------------
// Small helper kernels
// ---------------------------------------------------------------------------
__global__ void k_transpose_bf16(const float* __restrict__ src,
                                 bf16* __restrict__ dst, int K, int N) {
  int idx = blockIdx.x * blockDim.x + threadIdx.x;
  if (idx >= K * N) return;
  int n = idx / K, k = idx - n * K;
  dst[(size_t)n * K + k] = f2bf(src[(size_t)k * N + n]);   // [K,N] -> [N,K]
}

__global__ void k_u_to_a1(const float* __restrict__ U, bf16* __restrict__ A1) {
  int idx = blockIdx.x * blockDim.x + threadIdx.x;
  if (idx >= NROWS * ND) return;
  int row = idx >> 8, k = idx & 255;
  A1[(size_t)row * K1 + k] = f2bf(U[idx]);
}

__global__ void k_r_to_a1(const float* __restrict__ r, bf16* __restrict__ A1) {
  int idx = blockIdx.x * blockDim.x + threadIdx.x;
  if (idx >= NROWS * NH) return;
  int row = idx >> 7, k = idx & 127, b = row >> 10;
  A1[(size_t)row * K1 + ND + k] = f2bf(r[b * NH + k]);
}

__global__ void k_init(const float* __restrict__ U, float* us, float* ue,
                       float* h, float* c) {
  int id = blockIdx.x * blockDim.x + threadIdx.x;
  if (id < NB * ND) {
    int b = id >> 8, k = id & 255;
    us[id] = U[((size_t)b * NT + 0) * ND + k];
    ue[id] = U[((size_t)b * NT + 1) * ND + k];
  }
  if (id < NB * NH) { h[id] = 0.0f; c[id] = 0.0f; }
}

// r = tanh([h | us | ue] @ WD + b) : [32, 640] @ [640, 128]
__global__ void k_r_tanh(const float* __restrict__ h, const float* __restrict__ us,
                         const float* __restrict__ ue, const float* __restrict__ WDw,
                         const float* __restrict__ WDb, float* __restrict__ r) {
  int id = blockIdx.x * blockDim.x + threadIdx.x;
  if (id >= NB * NH) return;
  int b = id >> 7, j = id & 127;
  float acc = WDb[j];
  for (int k = 0; k < NH; ++k) acc = fmaf(h[b * NH + k],  WDw[k * NH + j], acc);
  for (int k = 0; k < ND; ++k) acc = fmaf(us[b * ND + k], WDw[(NH + k) * NH + j], acc);
  for (int k = 0; k < ND; ++k) acc = fmaf(ue[b * ND + k], WDw[(NH + ND + k) * NH + j], acc);
  r[id] = tanhf(acc);
}

// score = maxpool16(concat(m1, m2) @ W3 + b3) : [32768, 256] @ [256, 16]
__global__ void k_m3_score(const float* __restrict__ m1, const float* __restrict__ m2,
                           const float* __restrict__ W3, const float* __restrict__ b3,
                           float* __restrict__ score) {
  int row = blockIdx.x * blockDim.x + threadIdx.x;
  if (row >= NROWS) return;
  float acc[16];
#pragma unroll
  for (int p = 0; p < 16; ++p) acc[p] = b3[p];
  const float* r1 = m1 + (size_t)row * NH;
  const float* r2 = m2 + (size_t)row * NH;
  for (int k = 0; k < NH; ++k) {
    float v1 = r1[k], v2 = r2[k];
    const float* w1 = W3 + k * 16;
    const float* w2 = W3 + (NH + k) * 16;
#pragma unroll
    for (int p = 0; p < 16; ++p) acc[p] = fmaf(v1, w1[p], fmaf(v2, w2[p], acc[p]));
  }
  float m = acc[0];
#pragma unroll
  for (int p = 1; p < 16; ++p) m = fmaxf(m, acc[p]);
  score[row] = m;
}

// Per-batch argmax over T plus gather of selected U row.
__global__ __launch_bounds__(256) void k_argmax_gather(
    const float* __restrict__ score, const float* __restrict__ U,
    float* __restrict__ sel, float* __restrict__ idx_out) {
  __shared__ float sv[256];
  __shared__ int   si[256];
  const int b = blockIdx.x, tid = threadIdx.x;
  const float* sc = score + (size_t)b * NT;
  float bv = -INFINITY; int bi = 0;
  for (int t = tid; t < NT; t += 256) {
    float v = sc[t];
    if (v > bv) { bv = v; bi = t; }
  }
  sv[tid] = bv; si[tid] = bi;
  __syncthreads();
  for (int s = 128; s > 0; s >>= 1) {
    if (tid < s) {
      float ov = sv[tid + s]; int oi = si[tid + s];
      if (ov > sv[tid] || (ov == sv[tid] && oi < si[tid])) { sv[tid] = ov; si[tid] = oi; }
    }
    __syncthreads();
  }
  const int best = si[0];
  if (tid == 0) idx_out[b] = (float)best;
  sel[(size_t)b * ND + tid] = U[((size_t)b * NT + best) * ND + tid];  // tid<256=ND
}

// gates = [us|ue] @ wih^T + bih + h @ whh^T + bhh
__global__ void k_lstm_gates(const float* __restrict__ us, const float* __restrict__ ue,
                             const float* __restrict__ h, const float* __restrict__ wih,
                             const float* __restrict__ whh, const float* __restrict__ bih,
                             const float* __restrict__ bhh, float* __restrict__ gates) {
  int id = blockIdx.x * blockDim.x + threadIdx.x;
  if (id >= NB * 4 * NH) return;
  int b = id >> 9, j = id & 511;
  float acc = bih[j] + bhh[j];
  const float* wr = wih + (size_t)j * 512;
  const float* u1 = us + (size_t)b * ND;
  const float* u2 = ue + (size_t)b * ND;
  for (int k = 0; k < ND; ++k) acc = fmaf(u1[k], wr[k], acc);
  for (int k = 0; k < ND; ++k) acc = fmaf(u2[k], wr[ND + k], acc);
  const float* hr = h + (size_t)b * NH;
  const float* w2 = whh + (size_t)j * NH;
  for (int k = 0; k < NH; ++k) acc = fmaf(hr[k], w2[k], acc);
  gates[id] = acc;
}

__global__ void k_lstm_update(const float* __restrict__ gates,
                              float* __restrict__ h, float* __restrict__ c) {
  int id = blockIdx.x * blockDim.x + threadIdx.x;
  if (id >= NB * NH) return;
  int b = id >> 7, j = id & 127;
  const float* g = gates + (size_t)b * 512;
  float i  = 1.0f / (1.0f + __expf(-g[j]));
  float f  = 1.0f / (1.0f + __expf(-g[j + 128]));
  float gg = tanhf(g[j + 256]);
  float o  = 1.0f / (1.0f + __expf(-g[j + 384]));
  float cn = f * c[id] + i * gg;
  c[id] = cn;
  h[id] = o * tanhf(cn);
}

// ---------------------------------------------------------------------------
extern "C" void kernel_launch(void* const* d_in, const int* in_sizes, int n_in,
                              void* d_out, int out_size, void* d_ws, size_t ws_size,
                              hipStream_t stream) {
  (void)in_sizes; (void)n_in; (void)out_size; (void)ws_size;

  const float* U      = (const float*)d_in[0];
  const float* WDs_w  = (const float*)d_in[1];
  const float* WDs_b  = (const float*)d_in[2];
  const float* W1s_w  = (const float*)d_in[3];
  const float* W1s_b  = (const float*)d_in[4];
  const float* W2s_w  = (const float*)d_in[5];
  const float* W2s_b  = (const float*)d_in[6];
  const float* W3s_w  = (const float*)d_in[7];
  const float* W3s_b  = (const float*)d_in[8];
  const float* WDe_w  = (const float*)d_in[9];
  const float* WDe_b  = (const float*)d_in[10];
  const float* W1e_w  = (const float*)d_in[11];
  const float* W1e_b  = (const float*)d_in[12];
  const float* W2e_w  = (const float*)d_in[13];
  const float* W2e_b  = (const float*)d_in[14];
  const float* W3e_w  = (const float*)d_in[15];
  const float* W3e_b  = (const float*)d_in[16];
  const float* l_wih  = (const float*)d_in[17];
  const float* l_whh  = (const float*)d_in[18];
  const float* l_bih  = (const float*)d_in[19];
  const float* l_bhh  = (const float*)d_in[20];

  char* ws = (char*)d_ws;
  size_t off = 0;
  auto alloc = [&](size_t bytes) -> void* {
    void* p = ws + off;
    off = (off + bytes + 255) & ~(size_t)255;
    return p;
  };

  bf16*  A1    = (bf16*) alloc((size_t)NROWS * K1 * 2);   // concat(U_bf16, r_bf16)
  bf16*  W1st  = (bf16*) alloc((size_t)NPRE * K1 * 2);    // W1s transposed bf16
  bf16*  W1et  = (bf16*) alloc((size_t)NPRE * K1 * 2);
  bf16*  W2st  = (bf16*) alloc((size_t)NPRE * NH * 2);
  bf16*  W2et  = (bf16*) alloc((size_t)NPRE * NH * 2);
  float* m1f   = (float*)alloc((size_t)NROWS * NH * 4);
  bf16*  m1b   = (bf16*) alloc((size_t)NROWS * NH * 2);
  float* m2f   = (float*)alloc((size_t)NROWS * NH * 4);
  float* rbuf  = (float*)alloc((size_t)NB * NH * 4);
  float* usb   = (float*)alloc((size_t)NB * ND * 4);
  float* ueb   = (float*)alloc((size_t)NB * ND * 4);
  float* hb    = (float*)alloc((size_t)NB * NH * 4);
  float* cb    = (float*)alloc((size_t)NB * NH * 4);
  float* gates = (float*)alloc((size_t)NB * 4 * NH * 4);

  float* out       = (float*)d_out;
  float* alpha_out = out;                 // [32]
  float* beta_out  = out + NB;            // [32]
  float* ent_out   = out + 2 * NB;        // [4][2][32][1024]

  // One-time (per launch) weight conversion / transposition + state init
  k_transpose_bf16<<<(K1 * NPRE + 255) / 256, 256, 0, stream>>>(W1s_w, W1st, K1, NPRE);
  k_transpose_bf16<<<(K1 * NPRE + 255) / 256, 256, 0, stream>>>(W1e_w, W1et, K1, NPRE);
  k_transpose_bf16<<<(NH * NPRE + 255) / 256, 256, 0, stream>>>(W2s_w, W2st, NH, NPRE);
  k_transpose_bf16<<<(NH * NPRE + 255) / 256, 256, 0, stream>>>(W2e_w, W2et, NH, NPRE);
  k_u_to_a1<<<(NROWS * ND + 255) / 256, 256, 0, stream>>>(U, A1);
  k_init<<<(NB * ND + 255) / 256, 256, 0, stream>>>(U, usb, ueb, hb, cb);

  const dim3 gg(NROWS / 128, NPRE / 128);
  for (int it = 0; it < NITER; ++it) {
    // ---- start pointer (alpha) ----
    k_r_tanh<<<(NB * NH + 255) / 256, 256, 0, stream>>>(hb, usb, ueb, WDs_w, WDs_b, rbuf);
    k_r_to_a1<<<(NROWS * NH) / 256, 256, 0, stream>>>(rbuf, A1);
    k_gemm_maxout<<<gg, 256, 0, stream>>>(A1, K1, W1st, K1, K1, W1s_b, m1f, m1b, NH);
    k_gemm_maxout<<<gg, 256, 0, stream>>>(m1b, NH, W2st, NH, NH, W2s_b, m2f, nullptr, NH);
    float* sc_a = ent_out + ((size_t)it * 2 + 0) * NROWS;
    k_m3_score<<<NROWS / 256, 256, 0, stream>>>(m1f, m2f, W3s_w, W3s_b, sc_a);
    k_argmax_gather<<<NB, 256, 0, stream>>>(sc_a, U, usb, alpha_out);

    // ---- end pointer (beta) ----
    k_r_tanh<<<(NB * NH + 255) / 256, 256, 0, stream>>>(hb, usb, ueb, WDe_w, WDe_b, rbuf);
    k_r_to_a1<<<(NROWS * NH) / 256, 256, 0, stream>>>(rbuf, A1);
    k_gemm_maxout<<<gg, 256, 0, stream>>>(A1, K1, W1et, K1, K1, W1e_b, m1f, m1b, NH);
    k_gemm_maxout<<<gg, 256, 0, stream>>>(m1b, NH, W2et, NH, NH, W2e_b, m2f, nullptr, NH);
    float* sc_b = ent_out + ((size_t)it * 2 + 1) * NROWS;
    k_m3_score<<<NROWS / 256, 256, 0, stream>>>(m1f, m2f, W3e_w, W3e_b, sc_b);
    k_argmax_gather<<<NB, 256, 0, stream>>>(sc_b, U, ueb, beta_out);

    // ---- LSTM state update ----
    k_lstm_gates<<<(NB * 512) / 256, 256, 0, stream>>>(usb, ueb, hb, l_wih, l_whh,
                                                       l_bih, l_bhh, gates);
    k_lstm_update<<<(NB * NH) / 256, 256, 0, stream>>>(gates, hb, cb);
  }
}
